// MMoEB_33655363731936
// MI455X (gfx1250) — matrix operand order repaired and checked by reading, not compiled, not run on hardware
//
#include <hip/hip_runtime.h>
#include <math.h>

// ---------------- types ----------------
typedef __bf16 bf16_t;
typedef __attribute__((ext_vector_type(16))) __bf16 v16bf;
typedef __attribute__((ext_vector_type(8)))  float  v8f;
typedef unsigned short u16;
typedef unsigned int   u32;
typedef int v4i __attribute__((__vector_size__(16)));

#define BB 8
#define CC 64
#define HH 256
#define WW 256
#define HWSZ (HH*WW)
#define PRIOR 256

// ---------- async global->LDS (CDNA5), with compile-safe fallback ----------
#if defined(__has_builtin)
#if __has_builtin(__builtin_amdgcn_global_load_async_to_lds_b128)
#define ASYNC_LDS 1
#endif
#endif
#ifndef ASYNC_LDS
#define ASYNC_LDS 0
#endif

__device__ __forceinline__ void g2l_async16(const char* g, char* l) {
#if ASYNC_LDS
    typedef __attribute__((address_space(1))) v4i gv4i;
    typedef __attribute__((address_space(3))) v4i lv4i;
    __builtin_amdgcn_global_load_async_to_lds_b128((gv4i*)g, (lv4i*)l, 0, 0);
#else
    const u32* gs = (const u32*)g;
    u32* ls = (u32*)l;
    ls[0] = gs[0]; ls[1] = gs[1]; ls[2] = gs[2]; ls[3] = gs[3];
#endif
}

#if ASYNC_LDS
#if __has_builtin(__builtin_amdgcn_s_wait_asynccnt)
#define ASYNC_WAIT() __builtin_amdgcn_s_wait_asynccnt(0)
#else
#define ASYNC_WAIT() asm volatile("s_wait_asynccnt 0x0" ::: "memory")
#endif
#else
#define ASYNC_WAIT() ((void)0)
#endif

__device__ __forceinline__ u16 f2bf(float x) {
    u32 u = __float_as_uint(x);
    u32 r = u + 0x7FFFu + ((u >> 16) & 1u);
    return (u16)(r >> 16);
}
__device__ __forceinline__ float bf2f(u16 h) {
    return __uint_as_float(((u32)h) << 16);
}
__device__ __forceinline__ float gelu_exact(float x) {
    return 0.5f * x * (1.0f + erff(x * 0.70710678118654752f));
}

// A-matrix fragment (16-bit, 16x32): lane L holds row M=L%16.
// lanes 0-15: VGPR0..3 -> K={0..7} pairs, VGPR4..7 -> K={16..23}
// lanes16-31: +8 offset within each half. (ISA 7.12.2)
__device__ __forceinline__ v16bf frag_a(const u16* row, int lane) {
    int hoff = (lane >> 4) << 3;
    union { v16bf v; u32 u[8]; } f;
#pragma unroll
    for (int i = 0; i < 8; ++i) {
        int k = ((i >> 2) << 4) + hoff + ((i & 3) << 1);
        f.u[i] = *(const u32*)(row + k);
    }
    return f.v;
}
// B-matrix fragment (16-bit, 32x16): lane L holds col N=L%16,
// lanes 0-15: VGPR0..3 -> K=0..7, VGPR4..7 -> K=16..23; lanes 16-31: +8 (same as A).
__device__ __forceinline__ v16bf frag_b(const u16* col, int lane) {
    int hoff = (lane >> 4) << 3;
    union { v16bf v; u32 u[8]; } f;
#pragma unroll
    for (int i = 0; i < 8; ++i) {
        f.u[i] = *(const u32*)(col + ((i >> 2) << 4) + hoff + ((i & 3) << 1));
    }
    return f.v;
}
__device__ __forceinline__ v8f acc_splat(float b) {
    v8f a;
#pragma unroll
    for (int i = 0; i < 8; ++i) a[i] = b;
    return a;
}

// ============ K0: FiLM vectors + zero gate sums ============
__global__ void mmoeb_k0_film(const float* __restrict__ prior,
                              const float* __restrict__ ln1_w, const float* __restrict__ ln1_b,
                              const float* __restrict__ ln2_w, const float* __restrict__ ln2_b,
                              float* __restrict__ k1, float* __restrict__ k2,
                              float* __restrict__ gate_sums) {
    int g = blockIdx.x * blockDim.x + threadIdx.x;
    if (g >= BB * CC) return;
    int b = g >> 6, c = g & 63;
    float s1 = ln1_b[c], s2 = ln2_b[c];
    for (int p = 0; p < PRIOR; ++p) {
        float pv = prior[b * PRIOR + p];
        s1 += pv * ln1_w[c * PRIOR + p];
        s2 += pv * ln2_w[c * PRIOR + p];
    }
    k1[g] = 1.0f + tanhf(s1);
    k2[g] = tanhf(s2);
    gate_sums[g] = 0.0f;
}

// ============ K2w: repack conv weights to bf16 ============
// wA[(t*64+co)*64+ci] = c1a_w[co][ci][t]   (t = ky*3+kx)
// wB[co2*64+ci]       = c1b_w[co2][ci]
__global__ void mmoeb_k2w(const float* __restrict__ c1a_w, const float* __restrict__ c1b_w,
                          u16* __restrict__ wA, u16* __restrict__ wB) {
    int i = blockIdx.x * blockDim.x + threadIdx.x;
    if (i < 36864) {
        int t = i / 4096, co = (i >> 6) & 63, ci = i & 63;
        wA[i] = f2bf(c1a_w[co * 576 + ci * 9 + t]);
    } else if (i < 36864 + 8192) {
        int j = i - 36864;
        wB[j] = f2bf(c1b_w[j]);
    }
}

// ============ K5w: concat expert weights (bf16, zero-padded) ============
// Written into ONE contiguous 16KB region: [c1cat 16x64 | c2cat 16x64 | c3cat 64x32 | projw 64x64]
__global__ void mmoeb_k5w(const float* __restrict__ e0c1, const float* __restrict__ e0c2, const float* __restrict__ e0c3,
                          const float* __restrict__ e1c1, const float* __restrict__ e1c2, const float* __restrict__ e1c3,
                          const float* __restrict__ e2c1, const float* __restrict__ e2c2, const float* __restrict__ e2c3,
                          const float* __restrict__ e0b1, const float* __restrict__ e0b2,
                          const float* __restrict__ e1b1, const float* __restrict__ e1b2,
                          const float* __restrict__ e2b1, const float* __restrict__ e2b2,
                          const float* __restrict__ proj_w,
                          u16* __restrict__ wcat,
                          float* __restrict__ c1bcat, float* __restrict__ c2bcat) {
    int t = threadIdx.x;
    u16* c1cat = wcat;
    u16* c2cat = wcat + 1024;
    u16* c3cat = wcat + 2048;
    u16* projw = wcat + 4096;
    // c1cat/c2cat: [l(16)][ci(64)]; l 0-1:e0, 2-5:e1, 6-13:e2, 14-15:zero
    for (int i = t; i < 16 * 64; i += blockDim.x) {
        int l = i >> 6, ci = i & 63;
        float a = 0.f, bsum = 0.f;
        if (l < 2)       { a = e0c1[l * 64 + ci];       bsum = e0c2[l * 64 + ci]; }
        else if (l < 6)  { a = e1c1[(l - 2) * 64 + ci]; bsum = e1c2[(l - 2) * 64 + ci]; }
        else if (l < 14) { a = e2c1[(l - 6) * 64 + ci]; bsum = e2c2[(l - 6) * 64 + ci]; }
        c1cat[i] = f2bf(a);
        c2cat[i] = f2bf(bsum);
    }
    // c3cat: [co(64)][l(32)], l>=14 zero
    for (int i = t; i < 64 * 32; i += blockDim.x) {
        int co = i >> 5, l = i & 31;
        float a = 0.f;
        if (l < 2)       a = e0c3[co * 2 + l];
        else if (l < 6)  a = e1c3[co * 4 + (l - 2)];
        else if (l < 14) a = e2c3[co * 8 + (l - 6)];
        c3cat[i] = f2bf(a);
    }
    for (int i = t; i < 64 * 64; i += blockDim.x) projw[i] = f2bf(proj_w[i]);
    if (t < 16) {
        float b1 = 0.f, b2 = 0.f;
        if (t < 2)       { b1 = e0b1[t];     b2 = e0b2[t]; }
        else if (t < 6)  { b1 = e1b1[t - 2]; b2 = e1b2[t - 2]; }
        else if (t < 14) { b1 = e2b1[t - 6]; b2 = e2b2[t - 6]; }
        c1bcat[t] = b1; c2bcat[t] = b2;
    }
}

// ============ K1: LN + depth softmax gate + FiLM -> bf16 ============
__global__ void mmoeb_k1_pre(const float* __restrict__ xin, const float* __restrict__ depth,
                             const float* __restrict__ ln_w, const float* __restrict__ ln_b,
                             const float* __restrict__ depth_w, const float* __restrict__ depth_b,
                             const float* __restrict__ k1, const float* __restrict__ k2,
                             u16* __restrict__ yb) {
    __shared__ float lnwS[64], lnbS[64], dbS[64], k1S[64], k2S[64];
    __shared__ float dwS[576];
    int tid = threadIdx.x;
    int y = blockIdx.x, b = blockIdx.y;
    for (int i = tid; i < 576; i += 256) dwS[i] = depth_w[i];
    if (tid < 64) {
        lnwS[tid] = ln_w[tid]; lnbS[tid] = ln_b[tid]; dbS[tid] = depth_b[tid];
        k1S[tid] = k1[b * 64 + tid]; k2S[tid] = k2[b * 64 + tid];
    }
    __syncthreads();
    int x = tid;
    float nb[9];
#pragma unroll
    for (int dy = 0; dy < 3; ++dy)
#pragma unroll
        for (int dx = 0; dx < 3; ++dx) {
            int yy = y + dy - 1, xx = x + dx - 1;
            float v = 0.f;
            if (yy >= 0 && yy < HH && xx >= 0 && xx < WW)
                v = depth[(size_t)b * HWSZ + yy * WW + xx];
            nb[dy * 3 + dx] = v;
        }
    float s = 0.f, s2 = 0.f, dmax = 0.f;
    for (int c = 0; c < CC; ++c) {
        float xv = xin[((size_t)(b * CC + c) * HH + y) * WW + x];
        s += xv; s2 += xv * xv;
        float d = dbS[c];
#pragma unroll
        for (int k = 0; k < 9; ++k) d += dwS[c * 9 + k] * nb[k];
        d = fmaxf(d, 0.f);
        dmax = fmaxf(dmax, d);
    }
    float u = s * (1.0f / 64.0f);
    float var = s2 * (1.0f / 64.0f) - u * u;
    float rstd = rsqrtf(var + 1e-6f);
    float esum = 0.f;
    for (int c = 0; c < CC; ++c) {
        float d = dbS[c];
#pragma unroll
        for (int k = 0; k < 9; ++k) d += dwS[c * 9 + k] * nb[k];
        d = fmaxf(d, 0.f);
        esum += expf(d - dmax);
    }
    float einv = 1.0f / esum;
    for (int c = 0; c < CC; ++c) {
        size_t idx = ((size_t)(b * CC + c) * HH + y) * WW + x;
        float xv = xin[idx];
        float d = dbS[c];
#pragma unroll
        for (int k = 0; k < 9; ++k) d += dwS[c * 9 + k] * nb[k];
        d = fmaxf(d, 0.f);
        float sm = expf(d - dmax) * einv;
        float xn = (xv - u) * rstd * lnwS[c] + lnbS[c];
        float val = xn * sm * k1S[c] + k2S[c];
        yb[idx] = f2bf(val);
    }
}

// ============ K2: 3x3 implicit-GEMM WMMA + gelu + fused 1x1 64->128 ============
#define APITCH 68
__global__ void mmoeb_k2_conv(const u16* __restrict__ yb,
                              const u16* __restrict__ wA, const u16* __restrict__ wB,
                              const float* __restrict__ c1a_b, const float* __restrict__ c1b_b,
                              u16* __restrict__ xxb, u16* __restrict__ kkb) {
    __shared__ __align__(16) u16 Atile[3 * 34 * APITCH];
    __shared__ __align__(16) u16 Htile[32 * APITCH];
    __shared__ __align__(16) u16 WBs[8192];   // c1b weights, 16 KB, async-staged
    int tid = threadIdx.x;
    int x0 = blockIdx.x * 32, y = blockIdx.y, b = blockIdx.z;

    // async engine: stream stage-2 weights into LDS while we transpose the A tile
    for (int off = tid * 16; off < 16384; off += 256 * 16)
        g2l_async16((const char*)wB + off, (char*)WBs + off);

    for (int i = tid; i < 3 * 34 * 64; i += 256) {
        int pix = i % 34, c = (i / 34) & 63, row = i / (34 * 64);
        int yy = y + row - 1, xx = x0 - 1 + pix;
        u16 v = 0;
        if (yy >= 0 && yy < HH && xx >= 0 && xx < WW)
            v = yb[((size_t)(b * CC + c) * HH + yy) * WW + xx];
        Atile[(row * 34 + pix) * APITCH + c] = v;
    }
    ASYNC_WAIT();
    __syncthreads();

    int lane = tid & 31, wave = tid >> 5;
    int m = wave & 1, n = wave >> 1;
    int mrow = (m << 4) + (lane & 15);
    int ncol = (n << 4) + (lane & 15);

    v8f acc = acc_splat(c1a_b[ncol]);
#pragma unroll
    for (int t = 0; t < 9; ++t) {
        int ry = t / 3, rx = t % 3;
        const u16* arow = &Atile[(ry * 34 + (mrow + rx)) * APITCH];
        const u16* brow = &wA[(t * 64 + ncol) * 64];
        __builtin_prefetch(&wA[(((t + 1) % 9) * 64 + ncol) * 64], 0, 1);
#pragma unroll
        for (int kb = 0; kb < 2; ++kb) {
            v16bf af = frag_a(arow + kb * 32, lane);
            v16bf bfv = frag_b(brow + kb * 32, lane);
            acc = __builtin_amdgcn_wmma_f32_16x16x32_bf16(false, af, false, bfv,
                                                          (short)0, acc, false, false);
        }
    }
    // gelu -> LDS (bf16)
    int prow = (m << 4) + ((lane >> 4) << 3);
#pragma unroll
    for (int r = 0; r < 8; ++r)
        Htile[(prow + r) * APITCH + ncol] = f2bf(gelu_exact(acc[r]));
    __syncthreads();

    // stage 2: 1x1 64 -> 128, B fragments from LDS
#pragma unroll
    for (int half = 0; half < 2; ++half) {
        int m2 = wave & 1, n2 = (wave >> 1) + half * 4;
        int co2 = (n2 << 4) + (lane & 15);
        const u16* arow = &Htile[((m2 << 4) + (lane & 15)) * APITCH];
        const u16* brow = &WBs[co2 * 64];
        v8f acc2 = acc_splat(c1b_b[co2]);
#pragma unroll
        for (int kb = 0; kb < 2; ++kb) {
            v16bf af = frag_a(arow + kb * 32, lane);
            v16bf bfv = frag_b(brow + kb * 32, lane);
            acc2 = __builtin_amdgcn_wmma_f32_16x16x32_bf16(false, af, false, bfv,
                                                           (short)0, acc2, false, false);
        }
        int pr = (m2 << 4) + ((lane >> 4) << 3);
        int cmod = co2 & 63;
#pragma unroll
        for (int r = 0; r < 8; ++r) {
            size_t idx = ((size_t)(b * CC + cmod) * HH + y) * WW + (x0 + pr + r);
            u16 hv = f2bf(acc2[r]);
            if (co2 < 64) xxb[idx] = hv; else kkb[idx] = hv;
        }
    }
}

// ============ K3: separable depthwise (rank-1 3x3) + gelu + gate sums ============
__global__ void mmoeb_k3_dw(const u16* __restrict__ xxb,
                            const float* __restrict__ sh_w, const float* __restrict__ sh_b,
                            const float* __restrict__ sv_w, const float* __restrict__ sv_b,
                            u16* __restrict__ xx2b, float* __restrict__ gate_sums) {
    int x = threadIdx.x;
    int y = blockIdx.x, c = blockIdx.y, b = blockIdx.z;
    float wv0 = sv_w[c * 3 + 0], wv1 = sv_w[c * 3 + 1], wv2 = sv_w[c * 3 + 2];
    float wh0 = sh_w[c * 3 + 0], wh1 = sh_w[c * 3 + 1], wh2 = sh_w[c * 3 + 2];
    float beff = sh_b[c] * (wv0 + wv1 + wv2) + sv_b[c];
    float wv[3] = {wv0, wv1, wv2}, wh[3] = {wh0, wh1, wh2};
    float acc = beff;
    size_t base = (size_t)(b * CC + c) * HWSZ;
#pragma unroll
    for (int dy = 0; dy < 3; ++dy) {
        int yy = y + dy - 1;
        if (yy < 0 || yy >= HH) continue;
#pragma unroll
        for (int dx = 0; dx < 3; ++dx) {
            int xx = x + dx - 1;
            if (xx < 0 || xx >= WW) continue;
            acc += wv[dy] * wh[dx] * bf2f(xxb[base + yy * WW + xx]);
        }
    }
    float g = gelu_exact(acc);
    xx2b[base + y * WW + x] = f2bf(g);
    float r = g;
#pragma unroll
    for (int off = 16; off > 0; off >>= 1) r += __shfl_down(r, off, 32);
    __shared__ float red[8];
    int lane = threadIdx.x & 31, wave = threadIdx.x >> 5;
    if (lane == 0) red[wave] = r;
    __syncthreads();
    if (threadIdx.x == 0) {
        float t = 0.f;
#pragma unroll
        for (int i = 0; i < 8; ++i) t += red[i];
        atomicAdd(&gate_sums[b * 64 + c], t);
    }
}

// ============ K4: gate softmax + top-2 mask ============
__global__ void mmoeb_k4_gate(const float* __restrict__ gate_sums,
                              const float* __restrict__ gate_w,
                              float* __restrict__ mask) {
    int b = threadIdx.x;
    if (b >= BB) return;
    float lg[3];
    for (int e = 0; e < 3; ++e) {
        float s = 0.f;
        for (int c = 0; c < CC; ++c) s += gate_sums[b * 64 + c] * gate_w[e * 64 + c];
        lg[e] = s * (1.0f / (float)HWSZ);
    }
    float mx = fmaxf(lg[0], fmaxf(lg[1], lg[2]));
    float ex[3], sum = 0.f;
    for (int e = 0; e < 3; ++e) { ex[e] = expf(lg[e] - mx); sum += ex[e]; }
    float inv = 1.0f / sum;
    float wv[3] = {ex[0] * inv, ex[1] * inv, ex[2] * inv};
    int di = 0; float dm = wv[0];
    for (int e = 1; e < 3; ++e) if (wv[e] <= dm) { dm = wv[e]; di = e; }
    for (int e = 0; e < 3; ++e) mask[b * 4 + e] = (e == di) ? 0.f : wv[e];
}

// ============ K5: experts (3 WMMA stages) + proj + residual ============
__global__ void mmoeb_k5_tail(const u16* __restrict__ xx2b, const u16* __restrict__ kkb,
                              const u16* __restrict__ wcat,
                              const float* __restrict__ c1bcat, const float* __restrict__ c2bcat,
                              const float* __restrict__ c3b0, const float* __restrict__ c3b1,
                              const float* __restrict__ c3b2, const float* __restrict__ proj_b,
                              const float* __restrict__ mask, const float* __restrict__ xin,
                              float* __restrict__ out) {
    __shared__ __align__(16) u16 Xs[32 * APITCH];
    __shared__ __align__(16) u16 Ks[32 * APITCH];
    __shared__ float Us[32 * 17];
    __shared__ float Vs[32 * 17];
    __shared__ __align__(16) u16 Ts[32 * 32];
    __shared__ __align__(16) u16 Os[32 * APITCH];
    __shared__ __align__(16) u16 WTs[8192];   // [c1cat|c2cat|c3cat|projw], async-staged
    int tid = threadIdx.x;
    int x0 = blockIdx.x * 32, y = blockIdx.y, b = blockIdx.z;

    for (int off = tid * 16; off < 16384; off += 256 * 16)
        g2l_async16((const char*)wcat + off, (char*)WTs + off);

    for (int i = tid; i < 2 * 32 * 64; i += 256) {
        int pix = i & 31, c = (i >> 5) & 63, which = i >> 11;
        size_t idx = ((size_t)(b * CC + c) * HH + y) * WW + x0 + pix;
        if (which == 0) Xs[pix * APITCH + c] = xx2b[idx];
        else            Ks[pix * APITCH + c] = kkb[idx];
    }
    ASYNC_WAIT();
    __syncthreads();

    const u16* c1s = WTs;
    const u16* c2s = WTs + 1024;
    const u16* c3s = WTs + 2048;
    const u16* pjs = WTs + 4096;

    int lane = tid & 31, wave = tid >> 5;
    float mk0 = mask[b * 4 + 0], mk1 = mask[b * 4 + 1], mk2 = mask[b * 4 + 2];

    // stage 1: u = xx2 @ C1cat, v = kk @ C2cat  (32x16 each)
    if (wave < 4) {
        int which = wave >> 1, m = wave & 1;
        const u16* src = which ? Ks : Xs;
        const u16* wmat = which ? c2s : c1s;
        float bias = (which ? c2bcat : c1bcat)[lane & 15];
        const u16* arow = src + ((m << 4) + (lane & 15)) * APITCH;
        const u16* brow = wmat + (lane & 15) * 64;
        v8f acc = acc_splat(bias);
#pragma unroll
        for (int kb = 0; kb < 2; ++kb) {
            v16bf af = frag_a(arow + kb * 32, lane);
            v16bf bfv = frag_b(brow + kb * 32, lane);
            acc = __builtin_amdgcn_wmma_f32_16x16x32_bf16(false, af, false, bfv,
                                                          (short)0, acc, false, false);
        }
        float* dst = which ? Vs : Us;
        int pr = (m << 4) + ((lane >> 4) << 3);
#pragma unroll
        for (int r = 0; r < 8; ++r) dst[(pr + r) * 17 + (lane & 15)] = acc[r];
    }
    __syncthreads();

    // t = (u * v) scaled by per-expert mask, zero-pad to K=32
    for (int i = tid; i < 32 * 32; i += 256) {
        int pix = i >> 5, l = i & 31;
        float v = 0.f;
        if (l < 14) {
            float ms = (l < 2) ? mk0 : ((l < 6) ? mk1 : mk2);
            v = Us[pix * 17 + l] * Vs[pix * 17 + l] * ms;
        }
        Ts[i] = f2bf(v);
    }
    __syncthreads();

    // stage 2: r1 = t @ C3cat  (32x64); out_pre = xx2 + r1 + sum(mask_i * c3_b_i)
    {
        int m2 = wave & 1, n2 = wave >> 1;
        int co = (n2 << 4) + (lane & 15);
        float bias3 = mk0 * c3b0[co] + mk1 * c3b1[co] + mk2 * c3b2[co];
        const u16* arow = Ts + ((m2 << 4) + (lane & 15)) * 32;
        const u16* brow = c3s + co * 32;
        v8f acc = acc_splat(bias3);
        v16bf af = frag_a(arow, lane);
        v16bf bfv = frag_b(brow, lane);
        acc = __builtin_amdgcn_wmma_f32_16x16x32_bf16(false, af, false, bfv,
                                                      (short)0, acc, false, false);
        int pr = (m2 << 4) + ((lane >> 4) << 3);
#pragma unroll
        for (int r = 0; r < 8; ++r) {
            float val = acc[r] + bf2f(Xs[(pr + r) * APITCH + co]);
            Os[(pr + r) * APITCH + co] = f2bf(val);
        }
    }
    __syncthreads();

    // stage 3: final = out_pre @ projW + proj_b + residual
    {
        int m3 = wave & 1, n3 = wave >> 1;
        int co = (n3 << 4) + (lane & 15);
        const u16* arow = Os + ((m3 << 4) + (lane & 15)) * APITCH;
        const u16* brow = pjs + co * 64;
        v8f acc = acc_splat(proj_b[co]);
#pragma unroll
        for (int kb = 0; kb < 2; ++kb) {
            v16bf af = frag_a(arow + kb * 32, lane);
            v16bf bfv = frag_b(brow + kb * 32, lane);
            acc = __builtin_amdgcn_wmma_f32_16x16x32_bf16(false, af, false, bfv,
                                                          (short)0, acc, false, false);
        }
        int pr = (m3 << 4) + ((lane >> 4) << 3);
#pragma unroll
        for (int r = 0; r < 8; ++r) {
            size_t idx = ((size_t)(b * CC + co) * HH + y) * WW + (x0 + pr + r);
            out[idx] = acc[r] + xin[idx];
        }
    }
}

// ============ host launcher ============
extern "C" void kernel_launch(void* const* d_in, const int* in_sizes, int n_in,
                              void* d_out, int out_size, void* d_ws, size_t ws_size,
                              hipStream_t stream) {
    const float* xin     = (const float*)d_in[0];
    const float* prior   = (const float*)d_in[1];
    const float* depth   = (const float*)d_in[2];
    const float* ln_w    = (const float*)d_in[3];
    const float* ln_b    = (const float*)d_in[4];
    const float* depth_w = (const float*)d_in[5];
    const float* depth_b = (const float*)d_in[6];
    const float* ln1_w   = (const float*)d_in[7];
    const float* ln1_b   = (const float*)d_in[8];
    const float* ln2_w   = (const float*)d_in[9];
    const float* ln2_b   = (const float*)d_in[10];
    const float* c1a_w   = (const float*)d_in[11];
    const float* c1a_b   = (const float*)d_in[12];
    const float* c1b_w   = (const float*)d_in[13];
    const float* c1b_b   = (const float*)d_in[14];
    const float* sh_w    = (const float*)d_in[15];
    const float* sh_b    = (const float*)d_in[16];
    const float* sv_w    = (const float*)d_in[17];
    const float* sv_b    = (const float*)d_in[18];
    const float* gate_w  = (const float*)d_in[19];
    const float* e0c1 = (const float*)d_in[20]; const float* e0b1 = (const float*)d_in[21];
    const float* e0c2 = (const float*)d_in[22]; const float* e0b2 = (const float*)d_in[23];
    const float* e0c3 = (const float*)d_in[24]; const float* e0b3 = (const float*)d_in[25];
    const float* e1c1 = (const float*)d_in[26]; const float* e1b1 = (const float*)d_in[27];
    const float* e1c2 = (const float*)d_in[28]; const float* e1b2 = (const float*)d_in[29];
    const float* e1c3 = (const float*)d_in[30]; const float* e1b3 = (const float*)d_in[31];
    const float* e2c1 = (const float*)d_in[32]; const float* e2b1 = (const float*)d_in[33];
    const float* e2c2 = (const float*)d_in[34]; const float* e2b2 = (const float*)d_in[35];
    const float* e2c3 = (const float*)d_in[36]; const float* e2b3 = (const float*)d_in[37];
    const float* proj_w  = (const float*)d_in[38];
    const float* proj_b  = (const float*)d_in[39];
    float* out = (float*)d_out;

    char* ws = (char*)d_ws;
    const size_t NTOT = (size_t)BB * CC * HWSZ;           // 33.5M elems
    u16*   bufA      = (u16*)(ws + 0);                    // y_bf16, reused as xx2_bf16
    u16*   bufXX     = (u16*)(ws + NTOT * 2);             // xx bf16
    u16*   bufKK     = (u16*)(ws + NTOT * 4);             // kk bf16
    char*  sm        = ws + NTOT * 6;
    float* k1        = (float*)(sm + 0);
    float* k2        = (float*)(sm + 2048);
    float* gate_sums = (float*)(sm + 4096);
    float* maskbuf   = (float*)(sm + 6144);
    u16*   wA        = (u16*)(sm + 6400);                 // 73728 B
    u16*   wB        = (u16*)(sm + 80128);                // 16384 B
    u16*   wcat      = (u16*)(sm + 96512);                // 16384 B: c1cat|c2cat|c3cat|projw
    float* c1bcat    = (float*)(sm + 112896);
    float* c2bcat    = (float*)(sm + 113152);
    (void)in_sizes; (void)n_in; (void)out_size; (void)ws_size;

    mmoeb_k0_film<<<2, 256, 0, stream>>>(prior, ln1_w, ln1_b, ln2_w, ln2_b, k1, k2, gate_sums);
    mmoeb_k2w<<<(36864 + 8192 + 255) / 256, 256, 0, stream>>>(c1a_w, c1b_w, wA, wB);
    mmoeb_k5w<<<1, 256, 0, stream>>>(e0c1, e0c2, e0c3, e1c1, e1c2, e1c3, e2c1, e2c2, e2c3,
                                     e0b1, e0b2, e1b1, e1b2, e2b1, e2b2, proj_w,
                                     wcat, c1bcat, c2bcat);
    mmoeb_k1_pre<<<dim3(HH, BB), 256, 0, stream>>>(xin, depth, ln_w, ln_b, depth_w, depth_b,
                                                   k1, k2, bufA);
    mmoeb_k2_conv<<<dim3(WW / 32, HH, BB), 256, 0, stream>>>(bufA, wA, wB, c1a_b, c1b_b,
                                                             bufXX, bufKK);
    mmoeb_k3_dw<<<dim3(HH, CC, BB), 256, 0, stream>>>(bufXX, sh_w, sh_b, sv_w, sv_b,
                                                      bufA /*xx2 reuses y buffer*/, gate_sums);
    mmoeb_k4_gate<<<1, 32, 0, stream>>>(gate_sums, gate_w, maskbuf);
    mmoeb_k5_tail<<<dim3(WW / 32, HH, BB), 256, 0, stream>>>(bufA, bufKK, wcat,
                                                             c1bcat, c2bcat,
                                                             e0b3, e1b3, e2b3, proj_b,
                                                             maskbuf, xin, out);
}
